// RNN_59236188946512
// MI455X (gfx1250) — matrix operand order repaired and checked by reading, not compiled, not run on hardware
//
#include <hip/hip_runtime.h>

// ---- CDNA5 WMMA types -------------------------------------------------------
typedef __attribute__((ext_vector_type(16))) __bf16 v16bf;
typedef __attribute__((ext_vector_type(8)))  float  v8f;
typedef unsigned int u32x4 __attribute__((ext_vector_type(4)));

union FragU { u32x4 q[2]; v16bf v; };

// A-matrix 16x32 bf16 fragment: lane holds row m, K halves {0..7,16..23} or {8..15,24..31}
__device__ __forceinline__ v16bf loadA(const __bf16* p) {
    FragU u;
    u.q[0] = *reinterpret_cast<const u32x4*>(p);        // K = kbase+khalf .. +7
    u.q[1] = *reinterpret_cast<const u32x4*>(p + 16);   // K = kbase+16+khalf .. +7
    return u.v;
}
// B-matrix 32x16 bf16 fragment: lane holds column n, K halves {0..7,16..23} or {8..15,24..31}
__device__ __forceinline__ v16bf loadB(const __bf16* p) {
    FragU u;
    u.q[0] = *reinterpret_cast<const u32x4*>(p);
    u.q[1] = *reinterpret_cast<const u32x4*>(p + 16);
    return u.v;
}

__device__ __forceinline__ v8f wmma_bf16(v16bf a, v16bf b, v8f c) {
    return __builtin_amdgcn_wmma_f32_16x16x32_bf16(
        /*neg_a=*/false, a, /*neg_b=*/false, b,
        /*c_mod=*/(short)0, c, /*reuse_a=*/false, /*reuse_b=*/false);
}

// single-instruction ReLU: v_med3_f32(x, 0, FLT_MAX) == clamp(x, 0, FLT_MAX)
__device__ __forceinline__ float relu1(float x) {
    return __builtin_amdgcn_fmed3f(x, 0.0f, 3.402823466e+38f);
}

#define RNN_T     1000
#define RNN_TIN   800
#define RNN_F     5
#define RNN_H     128
#define RNN_BT    16      // batch rows per workgroup
#define RNN_K0    32      // x K padded to one WMMA

__global__ __launch_bounds__(256)
void rnn_persistent_kernel(const float* __restrict__ xg,     // [256,800,5]
                           const float* __restrict__ hs,     // [2,256,128]
                           const float* __restrict__ wih0,   // [128,5]
                           const float* __restrict__ whh0,   // [128,128]
                           const float* __restrict__ bih0,   // [128]
                           const float* __restrict__ bhh0,   // [128]
                           const float* __restrict__ wih1,   // [128,128]
                           const float* __restrict__ whh1,   // [128,128]
                           const float* __restrict__ bih1,   // [128]
                           const float* __restrict__ bhh1,   // [128]
                           const float* __restrict__ wout,   // [1,128]
                           const float* __restrict__ boutp,  // [1]
                           float* __restrict__ out)          // 204800 y + 65536 h
{
    // --- LDS: all weights (bf16) + hidden state resident for whole sequence ---
    __shared__ __align__(16) __bf16 lWih0[RNN_H * RNN_K0];   //  8 KB (zero-padded K)
    __shared__ __align__(16) __bf16 lWhh0[RNN_H * RNN_H];    // 32 KB
    __shared__ __align__(16) __bf16 lWih1[RNN_H * RNN_H];    // 32 KB
    __shared__ __align__(16) __bf16 lWhh1[RNN_H * RNN_H];    // 32 KB
    __shared__ __align__(16) __bf16 lh0[RNN_BT * RNN_H];     //  4 KB
    __shared__ __align__(16) __bf16 lh1[RNN_BT * RNN_H];     //  4 KB
    __shared__ __align__(16) __bf16 lx [RNN_BT * RNN_K0];    //  1 KB
    __shared__ float lbias0[RNN_H];
    __shared__ float lbias1[RNN_H];
    __shared__ float lwout [RNN_H];

    const int tid  = threadIdx.x;
    const int lane = tid & 31;
    const int nl   = lane & 15;          // fragment column / A row
    const int hi   = lane >> 4;          // lane half
    const int kh8  = hi * 8;             // A second-half K offset
    const int k16  = hi * 8;             // B K offset (same as A)
    const int rh8  = hi * 8;             // C/D row offset
    const int n0   = (tid >> 5) * 16;    // wave's N-tile (8 waves x 16 cols = 128)
    const int base = blockIdx.x * RNN_BT;

    // per-thread x-staging slots: (row xr0, col xc) and (row xr0+8, col xc)
    const int xc  = tid & 31;
    const int xr0 = tid >> 5;            // 0..7

    // --- one-time setup: convert weights to bf16 in LDS ----------------------
    for (int i = tid; i < RNN_H * RNN_H; i += 256) lWhh0[i] = (__bf16)whh0[i];
    for (int i = tid; i < RNN_H * RNN_H; i += 256) lWih1[i] = (__bf16)wih1[i];
    for (int i = tid; i < RNN_H * RNN_H; i += 256) lWhh1[i] = (__bf16)whh1[i];
    for (int i = tid; i < RNN_H * RNN_K0; i += 256) {
        int j = i >> 5, k = i & 31;
        lWih0[i] = (k < RNN_F) ? (__bf16)wih0[j * RNN_F + k] : (__bf16)0.0f;
    }
    for (int i = tid; i < RNN_BT * RNN_H; i += 256) {
        int r = i >> 7, j = i & 127;
        lh0[i] = (__bf16)hs[(base + r) * RNN_H + j];
        lh1[i] = (__bf16)hs[256 * RNN_H + (base + r) * RNN_H + j];
    }
    if (tid < RNN_H) {
        lbias0[tid] = bih0[tid] + bhh0[tid];
        lbias1[tid] = bih1[tid] + bhh1[tid];
        lwout[tid]  = wout[tid];
    }
    // stage x_0 (zero-padded to K=32)
    {
        float a = 0.0f, b = 0.0f;
        if (xc < RNN_F) {
            a = xg[(size_t)(base + xr0)     * (RNN_TIN * RNN_F) + xc];
            b = xg[(size_t)(base + xr0 + 8) * (RNN_TIN * RNN_F) + xc];
        }
        lx[xr0 * RNN_K0 + xc]       = (__bf16)a;
        lx[(xr0 + 8) * RNN_K0 + xc] = (__bf16)b;
    }
    const float bout = boutp[0];
    __syncthreads();

    float* yout  = out;                    // [256,800]
    float* hout0 = out + 256 * RNN_TIN;    // [256,128]
    float* hout1 = hout0 + 256 * RNN_H;    // [256,128]

    // --- 1000-step recurrence, entirely on-chip, x pipelined by one step -----
    for (int t = 0; t < RNN_T; ++t) {
        // prefetch x_{t+1} into registers: load issues here, latency hidden
        // under the layer-0 WMMA block; written to LDS after the WAR barrier.
        float xa = 0.0f, xb = 0.0f;
        if ((t + 1 < RNN_TIN) && (xc < RNN_F)) {
            xa = xg[(size_t)(base + xr0)     * (RNN_TIN * RNN_F) + (t + 1) * RNN_F + xc];
            xb = xg[(size_t)(base + xr0 + 8) * (RNN_TIN * RNN_F) + (t + 1) * RNN_F + xc];
        }

        // ---- layer 0: h0 = relu(x Wih0^T + h0 Whh0^T + b) -------------------
        v8f acc0;
        {
            float b0 = lbias0[n0 + nl];
            #pragma unroll
            for (int v = 0; v < 8; ++v) acc0[v] = b0;
            v16bf a = loadA(&lx[nl * RNN_K0 + kh8]);
            v16bf b = loadB(&lWih0[(n0 + nl) * RNN_K0 + k16]);
            acc0 = wmma_bf16(a, b, acc0);
            #pragma unroll
            for (int kt = 0; kt < 4; ++kt) {
                a = loadA(&lh0[nl * RNN_H + kt * 32 + kh8]);
                b = loadB(&lWhh0[(n0 + nl) * RNN_H + kt * 32 + k16]);
                acc0 = wmma_bf16(a, b, acc0);
            }
            #pragma unroll
            for (int v = 0; v < 8; ++v) acc0[v] = relu1(acc0[v]);
        }
        __syncthreads();   // all reads of old lh0 AND of lx (x_t) done
        #pragma unroll
        for (int v = 0; v < 8; ++v)
            lh0[(v + rh8) * RNN_H + n0 + nl] = (__bf16)acc0[v];
        // stage x_{t+1} into lx (safe: lx reads for step t completed above;
        // visibility for step t+1 covered by the barriers below)
        if (t + 1 < RNN_T) {
            lx[xr0 * RNN_K0 + xc]       = (__bf16)xa;
            lx[(xr0 + 8) * RNN_K0 + xc] = (__bf16)xb;
        }
        if (t == RNN_T - 1) {
            #pragma unroll
            for (int v = 0; v < 8; ++v)
                hout0[(size_t)(base + v + rh8) * RNN_H + n0 + nl] = acc0[v];
        }
        __syncthreads();   // new lh0 visible

        // ---- layer 1: h1 = relu(h0 Wih1^T + h1 Whh1^T + b) ------------------
        v8f acc1;
        {
            float b1 = lbias1[n0 + nl];
            #pragma unroll
            for (int v = 0; v < 8; ++v) acc1[v] = b1;
            #pragma unroll
            for (int kt = 0; kt < 4; ++kt) {
                v16bf a = loadA(&lh0[nl * RNN_H + kt * 32 + kh8]);
                v16bf b = loadB(&lWih1[(n0 + nl) * RNN_H + kt * 32 + k16]);
                acc1 = wmma_bf16(a, b, acc1);
            }
            #pragma unroll
            for (int kt = 0; kt < 4; ++kt) {
                v16bf a = loadA(&lh1[nl * RNN_H + kt * 32 + kh8]);
                v16bf b = loadB(&lWhh1[(n0 + nl) * RNN_H + kt * 32 + k16]);
                acc1 = wmma_bf16(a, b, acc1);
            }
            #pragma unroll
            for (int v = 0; v < 8; ++v) acc1[v] = relu1(acc1[v]);
        }
        __syncthreads();   // all reads of old lh1 done
        #pragma unroll
        for (int v = 0; v < 8; ++v)
            lh1[(v + rh8) * RNN_H + n0 + nl] = (__bf16)acc1[v];
        if (t == RNN_T - 1) {
            #pragma unroll
            for (int v = 0; v < 8; ++v)
                hout1[(size_t)(base + v + rh8) * RNN_H + n0 + nl] = acc1[v];
        }
        __syncthreads();   // new lh1 visible (and lx for t+1 visible)

        // ---- output: y = sigmoid(h1 . w_out + b_out), t < 800 only ----------
        if (t < RNN_TIN) {
            int r = tid >> 4, seg = tid & 15;
            float p = 0.0f;
            #pragma unroll
            for (int k = 0; k < 8; ++k)
                p += (float)lh1[r * RNN_H + seg * 8 + k] * lwout[seg * 8 + k];
            p += __shfl_down(p, 8, 16);
            p += __shfl_down(p, 4, 16);
            p += __shfl_down(p, 2, 16);
            p += __shfl_down(p, 1, 16);
            if (seg == 0) {
                float y = p + bout;
                yout[(size_t)(base + r) * RNN_TIN + t] = 1.0f / (1.0f + __expf(-y));
            }
        }
    }
}

extern "C" void kernel_launch(void* const* d_in, const int* in_sizes, int n_in,
                              void* d_out, int out_size, void* d_ws, size_t ws_size,
                              hipStream_t stream) {
    const float* x     = (const float*)d_in[0];
    const float* hs    = (const float*)d_in[1];
    const float* wih0  = (const float*)d_in[2];
    const float* whh0  = (const float*)d_in[3];
    const float* bih0  = (const float*)d_in[4];
    const float* bhh0  = (const float*)d_in[5];
    const float* wih1  = (const float*)d_in[6];
    const float* whh1  = (const float*)d_in[7];
    const float* bih1  = (const float*)d_in[8];
    const float* bhh1  = (const float*)d_in[9];
    const float* wout  = (const float*)d_in[10];
    const float* bout  = (const float*)d_in[11];
    float* out = (float*)d_out;

    // 256 batch rows / 16 rows per WG -> 16 persistent workgroups, 8 waves each
    rnn_persistent_kernel<<<dim3(16), dim3(256), 0, stream>>>(
        x, hs, wih0, whh0, bih0, bhh0, wih1, whh1, bih1, bhh1, wout, bout, out);
}
